// DualLayer_40802189312723
// MI455X (gfx1250) — matrix operand order repaired and checked by reading, not compiled, not run on hardware
//
#include <hip/hip_runtime.h>

#define BB 2
#define VV 50000
#define DD 64
#define RR 24
#define EE 800000
#define HH 4
#define NT (BB*VV)            // 100000 tokens
#define NTD ((size_t)NT*DD)   // 6,400,000 floats per big array

typedef __attribute__((ext_vector_type(16))) _Float16 v16h;
typedef __attribute__((ext_vector_type(8)))  float    v8f;

// ---------------------------------------------------------------------------
// Per-wave GEMM: O(16x64) = A(16x64 f16, LDS, row stride 64) @ W(64x64),
// where Wt is the transposed weight in LDS: Wt[n*64 + k] = W[k][n].
// Uses v_wmma_f32_16x16x32_f16: K split in 2, N split in 4.
// A layout (16-bit A 16x32): lane l holds row (l&15); half i -> k = kbase +
//   (l>=16 ? 16 : 0) + i.
// B layout (32x16): lane l holds col (l&15); half i -> k = kbase +
//   (l>=16 ? 16 : 0) + i  (contiguous in Wt row).
// C/D layout: lane l -> col (l&15); vgpr j -> row j + (l>=16 ? 8 : 0).
// ---------------------------------------------------------------------------
__device__ __forceinline__ void wave_gemm(const _Float16* A, const _Float16* Wt,
                                          v8f acc[4]) {
    int lane = threadIdx.x & 31;
    int r  = lane & 15;
    int hi = lane >> 4;
#pragma unroll
    for (int nt = 0; nt < 4; ++nt) {
        v8f c = {};
#pragma unroll
        for (int kt = 0; kt < 2; ++kt) {
            int k0 = kt * 32;
            v16h a, b;
            const _Float16* ap = A + r * 64 + k0 + hi * 16;
#pragma unroll
            for (int i = 0; i < 16; ++i) a[i] = ap[i];
            const _Float16* bp = Wt + (nt * 16 + r) * 64 + k0 + hi * 16;
#pragma unroll
            for (int i = 0; i < 16; ++i) b[i] = bp[i];
            c = __builtin_amdgcn_wmma_f32_16x16x32_f16(false, a, false, b,
                                                       (short)0, c, false, false);
        }
        acc[nt] = c;
    }
}

// Load 64x64 f32 weight (row-major [k][n]) transposed into LDS f16 Wt[n][k].
template <int BLK>
__device__ __forceinline__ void load_wT(const float* W, _Float16* Wt) {
    for (int i = threadIdx.x; i < 64 * 64; i += BLK) {
        int k = i >> 6, n = i & 63;
        Wt[n * 64 + k] = (_Float16)W[i];
    }
}

// ---------------------------------------------------------------------------
// Kernel 1: relation table + one-hot column sum
// ---------------------------------------------------------------------------
__global__ __launch_bounds__(256) void krel(const float* z, const float* W_z,
                                            const float* b_z, const float* W_vx1,
                                            float* relW, float* colsum) {
    __shared__ float zs[BB * DD];
    int tid = threadIdx.x;
    if (tid < BB * DD) zs[tid] = z[tid];
    __syncthreads();
    for (int o = tid; o < RR * BB * DD; o += 256) {
        int r = o >> 7, bd = o & 127, b = bd >> 6, d = bd & 63;
        float acc = b_z[r * 64 + d];
        for (int k = 0; k < 64; ++k)
            acc += zs[b * 64 + k] * W_z[k * (DD * RR) + r * 64 + d];
        relW[o] = acc;
    }
    if (tid < 64) {
        float s = 0.f;
        for (int k = 0; k < 64; ++k) s += W_vx1[(64 + k) * 64 + tid];
        colsum[tid] = s;
    }
}

// ---------------------------------------------------------------------------
// Kernel 2: vx = relu(x@W1a + b1 + onehot*colsum) @ W2 + b2   (WMMA)
// ---------------------------------------------------------------------------
__global__ __launch_bounds__(256) void kmpnn(const float* x, const float* W_vx1,
                                             const float* b_vx1, const float* W_vx2,
                                             const float* b_vx2, const float* colsum,
                                             const int* h_index, float* vx) {
    __shared__ _Float16 Xh[128 * 64];
    __shared__ _Float16 Wt[64 * 64];
    __shared__ float sb[64], sc[64];
    int tid = threadIdx.x;
    int t0 = blockIdx.x * 128;
    int wv = tid >> 5, lane = tid & 31, hi = lane >> 4, cb = lane & 15, rb = wv * 16;

    for (int i = tid; i < 128 * 64; i += 256) {
        int gt = t0 + (i >> 6);
        Xh[i] = (gt < NT) ? (_Float16)x[(size_t)gt * 64 + (i & 63)] : (_Float16)0.f;
    }
    load_wT<256>(W_vx1, Wt);  // rows 0..63 of W_vx1
    if (tid < 64) { sb[tid] = b_vx1[tid]; sc[tid] = colsum[tid]; }
    __syncthreads();

    int ht0 = h_index[0];
    int ht1 = VV + h_index[1];

    v8f acc[4];
    wave_gemm(Xh + rb * 64, Wt, acc);
    __syncthreads();  // all waves done reading Wt
#pragma unroll
    for (int nt = 0; nt < 4; ++nt) {
        int n = nt * 16 + cb;
        float bv = sb[n], cv = sc[n];
#pragma unroll
        for (int j = 0; j < 8; ++j) {
            int row = rb + hi * 8 + j;
            int gt = t0 + row;
            float v = acc[nt][j] + bv + ((gt == ht0 || gt == ht1) ? cv : 0.f);
            Xh[row * 64 + n] = (_Float16)fmaxf(v, 0.f);
        }
    }
    load_wT<256>(W_vx2, Wt);
    if (tid < 64) sb[tid] = b_vx2[tid];
    __syncthreads();

    wave_gemm(Xh + rb * 64, Wt, acc);
#pragma unroll
    for (int nt = 0; nt < 4; ++nt) {
        int n = nt * 16 + cb;
        float bv = sb[n];
#pragma unroll
        for (int j = 0; j < 8; ++j) {
            int gt = t0 + rb + hi * 8 + j;
            if (gt < NT) vx[(size_t)gt * 64 + n] = acc[nt][j] + bv;
        }
    }
}

// ---------------------------------------------------------------------------
// Kernel 3: edge scatter  agg[b,dst,d] += relation[rel][b*64+d] * vx[b,src,d]
// One wave per edge; relation table in LDS; f32 atomics into L2-resident agg.
// ---------------------------------------------------------------------------
__global__ __launch_bounds__(256) void kedge(const float* vx, const float* relW,
                                             const int* src, const int* dst,
                                             const int* rel, float* agg) {
    __shared__ float rl[RR * 128];
    int tid = threadIdx.x;
    for (int i = tid; i < RR * 128; i += 256) rl[i] = relW[i];
    __syncthreads();
    int e = blockIdx.x * 8 + (tid >> 5);
    if (e >= EE) return;
    int s = src[e], d = dst[e], r = rel[e];
    int lane = tid & 31;
    int b = lane >> 4;
    int q = (lane & 15) * 4;
    size_t soff = ((size_t)b * VV + s) * 64 + q;
    size_t doff = ((size_t)b * VV + d) * 64 + q;
    const float* rp = &rl[r * 128 + b * 64 + q];
#pragma unroll
    for (int i = 0; i < 4; ++i)
        atomicAdd(&agg[doff + i], vx[soff + i] * rp[i]);
}

// ---------------------------------------------------------------------------
// Kernel 4: DualV tail + attention projections + kvs/ksum/vsum reduction
// qnout aliases agg, valout aliases vx (same-row read-before-write per block).
// ---------------------------------------------------------------------------
__global__ __launch_bounds__(256) void kdual(
    const float* vx, const float* agg, const float* beta_p,
    const float* W_o1, const float* b_o1, const float* W_o2, const float* b_o2,
    const float* ln1_g, const float* ln1_b,
    const float* W_q, const float* b_q, const float* W_k, const float* b_k,
    const float* W_v, const float* b_v,
    float* h2out, float* qnout, float* valout,
    float* kvs, float* ksum, float* vsum) {
    __shared__ float F[128 * 64];
    __shared__ _Float16 Xh[128 * 64];
    __shared__ _Float16 Wt[64 * 64];
    __shared__ float s0[64], s1[64], s2[64], s3[64];
    int tid = threadIdx.x;
    int t0 = blockIdx.x * 128;
    int wv = tid >> 5, lane = tid & 31, hi = lane >> 4, cb = lane & 15, rb = wv * 16;
    v8f acc[4];

    if (tid < 64) { s0[tid] = beta_p[tid]; s1[tid] = b_o1[tid]; }
    load_wT<256>(W_o1, Wt);
    __syncthreads();

    // h = agg + beta * vx
    for (int i = tid; i < 128 * 64; i += 256) {
        int gt = t0 + (i >> 6);
        int n = i & 63;
        float v = 0.f;
        if (gt < NT)
            v = agg[(size_t)gt * 64 + n] + s0[n] * vx[(size_t)gt * 64 + n];
        Xh[i] = (_Float16)v;
    }
    __syncthreads();

    // fc_out layer 1 (relu)
    wave_gemm(Xh + rb * 64, Wt, acc);
    __syncthreads();
#pragma unroll
    for (int nt = 0; nt < 4; ++nt) {
        int n = nt * 16 + cb;
        float bv = s1[n];
#pragma unroll
        for (int j = 0; j < 8; ++j) {
            int row = rb + hi * 8 + j;
            Xh[row * 64 + n] = (_Float16)fmaxf(acc[nt][j] + bv, 0.f);
        }
    }
    load_wT<256>(W_o2, Wt);
    if (tid < 64) { s1[tid] = b_o2[tid]; s2[tid] = ln1_g[tid]; s3[tid] = ln1_b[tid]; }
    __syncthreads();

    // fc_out layer 2 -> F
    wave_gemm(Xh + rb * 64, Wt, acc);
#pragma unroll
    for (int nt = 0; nt < 4; ++nt) {
        int n = nt * 16 + cb;
        float bv = s1[n];
#pragma unroll
        for (int j = 0; j < 8; ++j) F[(rb + hi * 8 + j) * 64 + n] = acc[nt][j] + bv;
    }
    __syncthreads();

    // LN + residual -> h2 (global + Xh as f16 gemm input)
    if (tid < 128) {
        int gt = t0 + tid;
        float* fr = &F[tid * 64];
        float mu = 0.f;
        for (int n = 0; n < 64; ++n) mu += fr[n];
        mu *= (1.f / 64.f);
        float var = 0.f;
        for (int n = 0; n < 64; ++n) { float d = fr[n] - mu; var += d * d; }
        var *= (1.f / 64.f);
        float inv = rsqrtf(var + 1e-5f);
        for (int n = 0; n < 64; ++n) {
            float hv = (fr[n] - mu) * inv * s2[n] + s3[n];
            if (gt < NT) {
                hv += vx[(size_t)gt * 64 + n];
                h2out[(size_t)gt * 64 + n] = hv;
            } else hv = 0.f;
            Xh[tid * 64 + n] = (_Float16)hv;
        }
    }
    load_wT<256>(W_q, Wt);
    if (tid < 64) s1[tid] = b_q[tid];
    __syncthreads();

    // Q projection -> F, per-head l2norm -> qnout
    wave_gemm(Xh + rb * 64, Wt, acc);
    __syncthreads();
#pragma unroll
    for (int nt = 0; nt < 4; ++nt) {
        int n = nt * 16 + cb;
        float bv = s1[n];
#pragma unroll
        for (int j = 0; j < 8; ++j) F[(rb + hi * 8 + j) * 64 + n] = acc[nt][j] + bv;
    }
    __syncthreads();
    if (tid < 128) {
        int gt = t0 + tid;
        if (gt < NT) {
            float* fr = &F[tid * 64];
            for (int h = 0; h < HH; ++h) {
                float ss = 0.f;
                for (int i = 0; i < 16; ++i) { float q = fr[h * 16 + i]; ss += q * q; }
                float inv = 1.f / fmaxf(sqrtf(ss), 1e-12f);
                for (int i = 0; i < 16; ++i)
                    qnout[(size_t)gt * 64 + h * 16 + i] = fr[h * 16 + i] * inv;
            }
        }
    }
    load_wT<256>(W_v, Wt);
    if (tid < 64) s1[tid] = b_v[tid];
    __syncthreads();

    // V projection -> F (val)
    wave_gemm(Xh + rb * 64, Wt, acc);
#pragma unroll
    for (int nt = 0; nt < 4; ++nt) {
        int n = nt * 16 + cb;
        float bv = s1[n];
#pragma unroll
        for (int j = 0; j < 8; ++j) F[(rb + hi * 8 + j) * 64 + n] = acc[nt][j] + bv;
    }
    __syncthreads();
    for (int i = tid; i < 128 * 64; i += 256) {
        int gt = t0 + (i >> 6);
        if (gt < NT) valout[(size_t)gt * 64 + (i & 63)] = F[i];
    }
    load_wT<256>(W_k, Wt);
    if (tid < 64) s1[tid] = b_k[tid];
    __syncthreads();

    // K projection -> Xh (raw f16), then per-head normalize in place
    wave_gemm(Xh + rb * 64, Wt, acc);
    __syncthreads();  // all waves done reading Xh(h2) before overwrite
#pragma unroll
    for (int nt = 0; nt < 4; ++nt) {
        int n = nt * 16 + cb;
        float bv = s1[n];
#pragma unroll
        for (int j = 0; j < 8; ++j)
            Xh[(rb + hi * 8 + j) * 64 + n] = (_Float16)(acc[nt][j] + bv);
    }
    __syncthreads();
    if (tid < 128) {
        for (int h = 0; h < HH; ++h) {
            float ss = 0.f;
            for (int i = 0; i < 16; ++i) { float k = (float)Xh[tid * 64 + h * 16 + i]; ss += k * k; }
            float inv = 1.f / fmaxf(sqrtf(ss), 1e-12f);
            for (int i = 0; i < 16; ++i)
                Xh[tid * 64 + h * 16 + i] = (_Float16)((float)Xh[tid * 64 + h * 16 + i] * inv);
        }
    }
    __syncthreads();

    // Block-local reduction of kvs / ksum / vsum, then one atomic per output.
    int valid = NT - t0; if (valid > 128) valid = 128;
    bool hasB0 = (t0 < VV);
    bool hasB1 = (t0 + valid > VV);
#pragma unroll
    for (int u = 0; u < 4; ++u) {
        int o = tid * 4 + u;         // 1024 outputs: h*256 + dk*16 + dv
        int h = o >> 8, dk = (o >> 4) & 15, dv = o & 15;
        float a0 = 0.f, a1 = 0.f;
        for (int t = 0; t < valid; ++t) {
            float kv = (float)Xh[t * 64 + h * 16 + dk] * F[t * 64 + h * 16 + dv];
            if (t0 + t < VV) a0 += kv; else a1 += kv;
        }
        if (hasB0) atomicAdd(&kvs[(0 * HH + h) * 256 + dk * 16 + dv], a0);
        if (hasB1) atomicAdd(&kvs[(1 * HH + h) * 256 + dk * 16 + dv], a1);
    }
    if (tid < 64) {  // ksum columns
        float a0 = 0.f, a1 = 0.f;
        for (int t = 0; t < valid; ++t) {
            float k = (float)Xh[t * 64 + tid];
            if (t0 + t < VV) a0 += k; else a1 += k;
        }
        if (hasB0) atomicAdd(&ksum[tid], a0);
        if (hasB1) atomicAdd(&ksum[64 + tid], a1);
    } else if (tid < 128) {  // vsum columns
        int c = tid - 64;
        float a0 = 0.f, a1 = 0.f;
        for (int t = 0; t < valid; ++t) {
            float v = F[t * 64 + c];
            if (t0 + t < VV) a0 += v; else a1 += v;
        }
        if (hasB0) atomicAdd(&vsum[c], a0);
        if (hasB1) atomicAdd(&vsum[64 + c], a1);
    }
}

// ---------------------------------------------------------------------------
// Kernel 5: attention finalize + LN + FFN (WMMA) + LN -> out
// 64 tokens / block, 128 threads (4 waves).
// ---------------------------------------------------------------------------
__global__ __launch_bounds__(128) void kfinal(
    const float* qn, const float* val, const float* h2,
    const float* kvs, const float* ksum, const float* vsum,
    const float* an_g, const float* an_b,
    const float* W_f1, const float* b_f1, const float* W_f2, const float* b_f2,
    const float* fn_g, const float* fn_b, float* out) {
    __shared__ float F[64 * 64];
    __shared__ _Float16 Xh[64 * 64];
    __shared__ _Float16 Wt[64 * 64];
    __shared__ float kvL[2048];
    __shared__ float ksL[128], vsL[128];
    __shared__ float sa[64], sb[64], s1[64];
    int tid = threadIdx.x;
    int t0 = blockIdx.x * 64;
    int wv = tid >> 5, lane = tid & 31, hi = lane >> 4, cb = lane & 15, rb = wv * 16;
    v8f acc[4];

    for (int i = tid; i < 2048; i += 128) kvL[i] = kvs[i];
    if (tid < 128) { ksL[tid] = ksum[tid]; vsL[tid] = vsum[tid]; }
    if (tid < 64) { sa[tid] = an_g[tid]; sb[tid] = an_b[tid]; }
    __syncthreads();

    // attention finalize + first LN, per token (threads 0..63)
    if (tid < 64) {
        int gt = t0 + tid;
        float* fr = &F[tid * 64];
        if (gt < NT) {
            int b = (gt >= VV) ? 1 : 0;
            float qv[64];
#pragma unroll
            for (int n = 0; n < 64; ++n) qv[n] = qn[(size_t)gt * 64 + n];
#pragma unroll
            for (int h = 0; h < HH; ++h) {
                float den = 2.0f * (float)VV;
#pragma unroll
                for (int dk = 0; dk < 16; ++dk)
                    den += qv[h * 16 + dk] * ksL[b * 64 + h * 16 + dk];
                float invden = 1.0f / den;
                const float* kvb = &kvL[(b * HH + h) * 256];
#pragma unroll
                for (int dv = 0; dv < 16; ++dv) {
                    float num = 0.f;
#pragma unroll
                    for (int dk = 0; dk < 16; ++dk)
                        num += qv[h * 16 + dk] * kvb[dk * 16 + dv];
                    float vvv = val[(size_t)gt * 64 + h * 16 + dv];
                    num += vsL[b * 64 + h * 16 + dv] + vvv * (float)VV;
                    fr[h * 16 + dv] = h2[(size_t)gt * 64 + h * 16 + dv] + num * invden;
                }
            }
            float mu = 0.f;
            for (int n = 0; n < 64; ++n) mu += fr[n];
            mu *= (1.f / 64.f);
            float var = 0.f;
            for (int n = 0; n < 64; ++n) { float d = fr[n] - mu; var += d * d; }
            var *= (1.f / 64.f);
            float inv = rsqrtf(var + 1e-5f);
            for (int n = 0; n < 64; ++n)
                fr[n] = (fr[n] - mu) * inv * sa[n] + sb[n];
        } else {
            for (int n = 0; n < 64; ++n) fr[n] = 0.f;
        }
    }
    __syncthreads();

    // FFN layer 1
    for (int i = tid; i < 64 * 64; i += 128) Xh[i] = (_Float16)F[i];
    load_wT<128>(W_f1, Wt);
    if (tid < 64) s1[tid] = b_f1[tid];
    __syncthreads();
    wave_gemm(Xh + rb * 64, Wt, acc);
    __syncthreads();
#pragma unroll
    for (int nt = 0; nt < 4; ++nt) {
        int n = nt * 16 + cb;
        float bv = s1[n];
#pragma unroll
        for (int j = 0; j < 8; ++j)
            Xh[(rb + hi * 8 + j) * 64 + n] = (_Float16)fmaxf(acc[nt][j] + bv, 0.f);
    }
    load_wT<128>(W_f2, Wt);
    if (tid < 64) { s1[tid] = b_f2[tid]; sa[tid] = fn_g[tid]; sb[tid] = fn_b[tid]; }
    __syncthreads();

    // FFN layer 2 + residual into F
    wave_gemm(Xh + rb * 64, Wt, acc);
#pragma unroll
    for (int nt = 0; nt < 4; ++nt) {
        int n = nt * 16 + cb;
        float bv = s1[n];
#pragma unroll
        for (int j = 0; j < 8; ++j)
            F[(rb + hi * 8 + j) * 64 + n] += acc[nt][j] + bv;
    }
    __syncthreads();

    // final LN -> out
    if (tid < 64) {
        int gt = t0 + tid;
        if (gt < NT) {
            float* fr = &F[tid * 64];
            float mu = 0.f;
            for (int n = 0; n < 64; ++n) mu += fr[n];
            mu *= (1.f / 64.f);
            float var = 0.f;
            for (int n = 0; n < 64; ++n) { float d = fr[n] - mu; var += d * d; }
            var *= (1.f / 64.f);
            float inv = rsqrtf(var + 1e-5f);
            for (int n = 0; n < 64; ++n)
                out[(size_t)gt * 64 + n] = (fr[n] - mu) * inv * sa[n] + sb[n];
        }
    }
}

// ---------------------------------------------------------------------------
extern "C" void kernel_launch(void* const* d_in, const int* in_sizes, int n_in,
                              void* d_out, int out_size, void* d_ws, size_t ws_size,
                              hipStream_t stream) {
    const float* x      = (const float*)d_in[0];
    const float* z      = (const float*)d_in[1];
    const float* W_vx1  = (const float*)d_in[2];
    const float* b_vx1  = (const float*)d_in[3];
    const float* W_vx2  = (const float*)d_in[4];
    const float* b_vx2  = (const float*)d_in[5];
    const float* W_z    = (const float*)d_in[6];
    const float* b_z    = (const float*)d_in[7];
    const float* beta_p = (const float*)d_in[8];
    const float* W_o1   = (const float*)d_in[9];
    const float* b_o1   = (const float*)d_in[10];
    const float* W_o2   = (const float*)d_in[11];
    const float* b_o2   = (const float*)d_in[12];
    const float* ln1_g  = (const float*)d_in[13];
    const float* ln1_b  = (const float*)d_in[14];
    const float* W_q    = (const float*)d_in[15];
    const float* b_q    = (const float*)d_in[16];
    const float* W_k    = (const float*)d_in[17];
    const float* b_k    = (const float*)d_in[18];
    const float* W_v    = (const float*)d_in[19];
    const float* b_v    = (const float*)d_in[20];
    const float* an_g   = (const float*)d_in[21];
    const float* an_b   = (const float*)d_in[22];
    const float* W_f1   = (const float*)d_in[23];
    const float* b_f1   = (const float*)d_in[24];
    const float* W_f2   = (const float*)d_in[25];
    const float* b_f2   = (const float*)d_in[26];
    const float* fn_g   = (const float*)d_in[27];
    const float* fn_b   = (const float*)d_in[28];
    const int* h_index  = (const int*)d_in[29];
    // d_in[30] = r_index (unused by reference)
    const int* src      = (const int*)d_in[31];
    const int* dst      = (const int*)d_in[32];
    const int* rel      = (const int*)d_in[33];

    float* ws     = (float*)d_ws;
    float* relW   = ws;            // 3072
    float* colsum = ws + 3072;     // 64
    float* kvs    = ws + 3136;     // 2048
    float* ksum   = ws + 5184;     // 128
    float* vsum   = ws + 5312;     // 128
    float* bigA   = ws + 8192;          // vx  -> later val
    float* bigB   = bigA + NTD;         // agg -> later qn
    float* bigC   = bigB + NTD;         // h2

    // zero agg and the small reduction buffers every call
    hipMemsetAsync(bigB, 0, NTD * sizeof(float), stream);
    hipMemsetAsync(kvs, 0, (2048 + 128 + 128) * sizeof(float), stream);

    krel<<<1, 256, 0, stream>>>(z, W_z, b_z, W_vx1, relW, colsum);
    kmpnn<<<(NT + 127) / 128, 256, 0, stream>>>(x, W_vx1, b_vx1, W_vx2, b_vx2,
                                                colsum, h_index, bigA);
    kedge<<<(EE + 7) / 8, 256, 0, stream>>>(bigA, relW, src, dst, rel, bigB);
    kdual<<<(NT + 127) / 128, 256, 0, stream>>>(
        bigA, bigB, beta_p, W_o1, b_o1, W_o2, b_o2, ln1_g, ln1_b,
        W_q, b_q, W_k, b_k, W_v, b_v,
        /*h2out=*/bigC, /*qnout=*/bigB, /*valout=*/bigA, kvs, ksum, vsum);
    kfinal<<<(NT + 63) / 64, 128, 0, stream>>>(
        /*qn=*/bigB, /*val=*/bigA, /*h2=*/bigC, kvs, ksum, vsum,
        an_g, an_b, W_f1, b_f1, W_f2, b_f2, fn_g, fn_b, (float*)d_out);
}
